// gtnet_30434138260167
// MI455X (gfx1250) — matrix operand order repaired and checked by reading, not compiled, not run on hardware
//
#include <hip/hip_runtime.h>
#include <math.h>

typedef _Float16 f16;
typedef __attribute__((ext_vector_type(16))) _Float16 v16h;
typedef __attribute__((ext_vector_type(8)))  float    v8f;

#define BB    16
#define NN    512
#define TT    168
#define CC    32
#define CSKIP 64
#define CEND  128
#define PRED  24
#define ND    40
#define KSUB  20
#define ALPHA 0.05f
#define EPSLN 1e-5f

// ---------------------------------------------------------------------------
// WMMA fragment swizzle helpers (CDNA5 16-bit A 16x32 / B 32x16 VGPR layouts)
// A layout [mTile][kTile][lane][16]: lane = halfSel*16+lrow (row m0+lrow);
//   element e<8  <-> K = kt + halfSel*8 + e
//   element e>=8 <-> K = kt + 16 + halfSel*8 + (e-8)
// B layout [batch][kTile][nTile][lane][16]: lane = halfSel*16 + (t&15);
//   element e<8  <-> K = kt + halfSel*8 + e
//   element e>=8 <-> K = kt + 16 + halfSel*8 + (e-8)
// ---------------------------------------------------------------------------
__device__ __forceinline__ int aswz(int v, int w) {
  int mTile = v >> 4, lrow = v & 15;
  int kTile = w >> 5, kr = w & 31;
  int halfSel = (kr >> 3) & 1;
  int e = (kr & 7) | ((kr & 16) >> 1);
  int lane = halfSel * 16 + lrow;
  return (((mTile * 16 + kTile) * 32) + lane) * 16 + e;
}

__device__ __forceinline__ long long bswz(long long batch, int w, int t, int nT16) {
  int kTile = w >> 5, kr = w & 31;
  int halfSel = (kr >> 3) & 1, e = (kr & 7) | ((kr & 16) >> 1);
  int lane = halfSel * 16 + (t & 15);
  int nTile = t >> 4;
  return ((((batch * 16 + kTile) * (long long)nT16 + nTile) * 32) + lane) * 16 + e;
}

// ---------------------------------------------------------------------------
// Graph construction
// ---------------------------------------------------------------------------
__global__ void nodevec_kernel(const float* __restrict__ emb, const float* __restrict__ lw,
                               const float* __restrict__ lb, float* __restrict__ nv) {
  int i = blockIdx.x;
  int j = threadIdx.x;
  if (j >= ND) return;
  float acc = lb[j];
  for (int k = 0; k < ND; ++k) acc += emb[i * ND + k] * lw[j * ND + k];
  nv[i * ND + j] = tanhf(3.0f * acc);
}

__global__ void amat_kernel(const float* __restrict__ nv1, const float* __restrict__ nv2,
                            float* __restrict__ dense) {
  int idx = blockIdx.x * blockDim.x + threadIdx.x;
  if (idx >= NN * NN) return;
  int i = idx / NN, j = idx % NN;
  float s = 0.f;
  for (int k = 0; k < ND; ++k)
    s += nv1[i * ND + k] * nv2[j * ND + k] - nv2[i * ND + k] * nv1[j * ND + k];
  float a = tanhf(3.0f * s);
  dense[idx] = a > 0.f ? a : 0.f;
}

// per-row top-K (K=20) sparsification; one block per row
__global__ void topk_kernel(const float* __restrict__ adj0, float* __restrict__ adp) {
  __shared__ float sv[NN];
  __shared__ float rv[256];
  __shared__ int   ri[256];
  int r = blockIdx.x, t = threadIdx.x;
  sv[t]       = adj0[r * NN + t];
  sv[t + 256] = adj0[r * NN + t + 256];
  adp[r * NN + t]       = 0.f;
  adp[r * NN + t + 256] = 0.f;
  __syncthreads();
  for (int k = 0; k < KSUB; ++k) {
    float v0 = sv[t], v1 = sv[t + 256];
    float v;  int ii;
    if (v1 > v0) { v = v1; ii = t + 256; } else { v = v0; ii = t; }
    rv[t] = v; ri[t] = ii;
    __syncthreads();
    for (int s = 128; s > 0; s >>= 1) {
      if (t < s) {
        if (rv[t + s] > rv[t] || (rv[t + s] == rv[t] && ri[t + s] < ri[t])) {
          rv[t] = rv[t + s]; ri[t] = ri[t + s];
        }
      }
      __syncthreads();
    }
    if (t == 0) {
      int j = ri[0];
      adp[r * NN + j] = sv[j];
      sv[j] = -1e30f;
    }
    __syncthreads();
  }
}

__global__ void adj_sums_kernel(const float* __restrict__ adp, float* __restrict__ sums,
                                int transposed) {
  int r = blockIdx.x;
  __shared__ float sh[256];
  float acc = 0.f;
  for (int j = threadIdx.x; j < NN; j += 256)
    acc += transposed ? adp[j * NN + r] : adp[r * NN + j];
  sh[threadIdx.x] = acc;
  __syncthreads();
  for (int s = 128; s > 0; s >>= 1) {
    if (threadIdx.x < s) sh[threadIdx.x] += sh[threadIdx.x + s];
    __syncthreads();
  }
  if (threadIdx.x == 0) sums[r] = sh[0] + 1.0f;  // +I contribution
}

// emits fwd/bwd normalized adjacency directly in WMMA-A fragment order
__global__ void normadj_f16_kernel(const float* __restrict__ adp,
                                   const float* __restrict__ rowsum,
                                   const float* __restrict__ colsum,
                                   f16* __restrict__ Afwd, f16* __restrict__ Abwd) {
  int idx = blockIdx.x * blockDim.x + threadIdx.x;
  if (idx >= NN * NN) return;
  int v = idx / NN, w = idx % NN;
  float d = (v == w) ? 1.f : 0.f;
  int si = aswz(v, w);
  Afwd[si] = (f16)((adp[v * NN + w] + d) / rowsum[v]);
  Abwd[si] = (f16)((adp[w * NN + v] + d) / colsum[v]);
}

// ---------------------------------------------------------------------------
// Front end
// ---------------------------------------------------------------------------
__global__ void start_kernel(const float* __restrict__ x, const float* __restrict__ sw,
                             const float* __restrict__ sb, const float* __restrict__ pos,
                             float* __restrict__ xh) {
  long long total = (long long)BB * CC * NN * TT;
  long long idx = (long long)blockIdx.x * blockDim.x + threadIdx.x;
  if (idx >= total) return;
  int t = (int)(idx % TT);
  long long r = idx / TT;
  int n = (int)(r % NN); r /= NN;
  int c = (int)(r % CC);
  int b = (int)(r / CC);
  float xv = x[((long long)b * NN + n) * TT + t];
  xh[idx] = sw[c] * xv + sb[c] + pos[c * TT + t];
}

__global__ void skip0_kernel(const float* __restrict__ x, const float* __restrict__ w,
                             const float* __restrict__ bias, float* __restrict__ skip) {
  int idx = blockIdx.x * blockDim.x + threadIdx.x;
  if (idx >= BB * CSKIP * NN) return;
  int n = idx % NN;
  int r = idx / NN;
  int o = r % CSKIP;
  int b = r / CSKIP;
  const float* xp = x + ((long long)b * NN + n) * TT;
  const float* wp = w + o * TT;
  float acc = bias[o];
  for (int t = 0; t < TT; ++t) acc += wp[t] * xp[t];
  skip[idx] = acc;
}

// ---------------------------------------------------------------------------
// Dilated inception + gating  (outs order: k=2,3,6,7, each 8 channels)
// ---------------------------------------------------------------------------
struct IncArgs {
  const float* fw[4]; const float* fb[4];
  const float* gw[4]; const float* gb[4];
};

__global__ void inception_kernel(const float* __restrict__ xin, float* __restrict__ xout,
                                 IncArgs p, int Tin, int Tout, int dil) {
  long long total = (long long)BB * CC * NN * Tout;
  long long idx = (long long)blockIdx.x * blockDim.x + threadIdx.x;
  if (idx >= total) return;
  int t = (int)(idx % Tout);
  long long r = idx / Tout;
  int n = (int)(r % NN); r /= NN;
  int c = (int)(r % CC);
  int b = (int)(r / CC);
  const int kks[4] = {2, 3, 6, 7};
  int g = c >> 3, oc = c & 7;
  int kk = kks[g];
  int off = (7 - kk) * dil;  // align to last Tout samples
  const float* fw = p.fw[g];
  const float* gw = p.gw[g];
  float fa = p.fb[g][oc], ga = p.gb[g][oc];
  for (int ci = 0; ci < CC; ++ci) {
    const float* xp  = xin + (((long long)b * CC + ci) * NN + n) * Tin + t + off;
    const float* fwp = fw + (oc * CC + ci) * kk;
    const float* gwp = gw + (oc * CC + ci) * kk;
    for (int j = 0; j < kk; ++j) {
      float xv = xp[j * dil];
      fa += fwp[j] * xv;
      ga += gwp[j] * xv;
    }
  }
  float fo = tanhf(fa);
  float go = 1.f / (1.f + expf(-ga));
  xout[idx] = fo * go;
}

// skip conv: full-time-length kernel, output length 1, accumulates into skip
__global__ void skipconv_kernel(const float* __restrict__ xin, const float* __restrict__ w,
                                const float* __restrict__ bias, float* __restrict__ skip, int Ti) {
  int idx = blockIdx.x * blockDim.x + threadIdx.x;
  if (idx >= BB * CSKIP * NN) return;
  int n = idx % NN;
  int r = idx / NN;
  int o = r % CSKIP;
  int b = r / CSKIP;
  float acc = bias[o];
  for (int c = 0; c < CC; ++c) {
    const float* xp = xin + (((long long)b * CC + c) * NN + n) * Ti;
    const float* wp = w + ((long long)o * CC + c) * Ti;
    for (int t = 0; t < Ti; ++t) acc += wp[t] * xp[t];
  }
  skip[idx] += acc;
}

// ---------------------------------------------------------------------------
// mixprop: WMMA graph GEMM   Y[bc, v, t] = sum_w A[v,w] * H[bc, w, t]
// Operands pre-swizzled into fragment order: each lane loads one contiguous
// 32-byte chunk per fragment (2x global_load_b128).
// ---------------------------------------------------------------------------
__global__ void tof16_pad_kernel(const float* __restrict__ src, f16* __restrict__ dst,
                                 int Ti, int Tp) {
  long long total = (long long)BB * CC * NN * Tp;
  long long idx = (long long)blockIdx.x * blockDim.x + threadIdx.x;
  if (idx >= total) return;
  int t = (int)(idx % Tp);
  long long r = idx / Tp;             // r = batch*NN + w
  int w = (int)(r % NN);
  long long batch = r / NN;
  f16 v = (t < Ti) ? (f16)src[r * Ti + t] : (f16)0.f;
  dst[bswz(batch, w, t, Tp >> 4)] = v;
}

__global__ __launch_bounds__(256) void nconv_wmma_kernel(
    const f16* __restrict__ As,  // swizzled adjacency fragments
    const f16* __restrict__ Hs,  // swizzled activation fragments
    float* __restrict__ Y,       // [BC, 512, Tp] f32, row-major
    int Tp) {
  const int lane  = threadIdx.x & 31;
  const int wave  = threadIdx.x >> 5;
  const int mTile = blockIdx.y * 8 + wave;   // 0..31
  const int nT0   = blockIdx.x * 2;          // two 16-col tiles per wave
  const int batch = blockIdx.z;              // 0..511 (b*C + c)
  const int nT16  = Tp >> 4;

  const f16* ap = As + (((mTile * 16) * 32 + lane) << 4);
  const f16* bp = Hs + (((((long long)batch * 16) * nT16 + nT0) * 32 + lane) << 4);
  const long long kstride = (long long)nT16 * (32 * 16);

  v8f acc0 = {}, acc1 = {};
#pragma unroll
  for (int k16 = 0; k16 < 16; ++k16) {
    v16h a  = *(const v16h*)(ap + k16 * (32 * 16));
    v16h b0 = *(const v16h*)(bp + k16 * kstride);
    v16h b1 = *(const v16h*)(bp + k16 * kstride + 32 * 16);
    acc0 = __builtin_amdgcn_wmma_f32_16x16x32_f16(
        false, a, false, b0, (short)0, acc0, false, false);
    acc1 = __builtin_amdgcn_wmma_f32_16x16x32_f16(
        false, a, false, b1, (short)0, acc1, false, false);
  }
  const int halfSel = lane >> 4, lrow = lane & 15;
  float* Yb = Y + (long long)batch * NN * Tp;
  const int n0 = nT0 * 16;
#pragma unroll
  for (int r = 0; r < 8; ++r) {
    int m = mTile * 16 + r + halfSel * 8;
    Yb[(long long)m * Tp + n0 + lrow]      = acc0[r];
    Yb[(long long)m * Tp + n0 + 16 + lrow] = acc1[r];
  }
}

// h_{k+1} = alpha * x0 + (1-alpha) * Y ; also re-emit swizzled f16 for next hop
__global__ void hop_combine_kernel(const float* __restrict__ x0, const float* __restrict__ Y,
                                   float* __restrict__ hout, f16* __restrict__ hh,
                                   int Ti, int Tp) {
  long long total = (long long)BB * CC * NN * Tp;
  long long idx = (long long)blockIdx.x * blockDim.x + threadIdx.x;
  if (idx >= total) return;
  int t = (int)(idx % Tp);
  long long r = idx / Tp;
  int w = (int)(r % NN);
  long long batch = r / NN;
  float v = 0.f;
  if (t < Ti) {
    v = ALPHA * x0[r * Ti + t] + (1.f - ALPHA) * Y[idx];
    hout[r * Ti + t] = v;
  }
  hh[bswz(batch, w, t, Tp >> 4)] = (f16)v;
}

// mlp over concat [h0,h1,h2] for both directions + residual
__global__ void mixmlp_kernel(const float* __restrict__ h0,
                              const float* __restrict__ h1f, const float* __restrict__ h2f,
                              const float* __restrict__ h1b, const float* __restrict__ h2b,
                              const float* __restrict__ g1w, const float* __restrict__ g1b,
                              const float* __restrict__ g2w, const float* __restrict__ g2b,
                              const float* __restrict__ residual, int Tprev, int Ti,
                              float* __restrict__ out) {
  long long total = (long long)BB * CC * NN * Ti;
  long long idx = (long long)blockIdx.x * blockDim.x + threadIdx.x;
  if (idx >= total) return;
  int t = (int)(idx % Ti);
  long long r = idx / Ti;
  int n = (int)(r % NN); r /= NN;
  int o = (int)(r % CC);
  int b = (int)(r / CC);
  float acc = g1b[o] + g2b[o];
  long long base = (((long long)b * CC) * NN + n) * Ti + t;
  const long long cstride = (long long)NN * Ti;
  for (int c = 0; c < CC; ++c) {
    long long ix = base + c * cstride;
    float v0 = h0[ix];
    acc += (g1w[o * 96 + c] + g2w[o * 96 + c]) * v0;
    acc += g1w[o * 96 + 32 + c] * h1f[ix] + g1w[o * 96 + 64 + c] * h2f[ix];
    acc += g2w[o * 96 + 32 + c] * h1b[ix] + g2w[o * 96 + 64 + c] * h2b[ix];
  }
  acc += residual[(((long long)b * CC + o) * NN + n) * Tprev + (Tprev - Ti) + t];
  out[idx] = acc;
}

// ---------------------------------------------------------------------------
// LayerNorm over (C,N,T) per batch element
// ---------------------------------------------------------------------------
__global__ void zero_kernel(float* __restrict__ p, int n) {
  int i = blockIdx.x * blockDim.x + threadIdx.x;
  if (i < n) p[i] = 0.f;
}

__global__ void ln_reduce_kernel(const float* __restrict__ xx, float* __restrict__ stats, int Ti) {
  int b = blockIdx.y;
  long long M = (long long)CC * NN * Ti;
  const float* xb = xx + (long long)b * M;
  float s1 = 0.f, s2 = 0.f;
  for (long long i = (long long)blockIdx.x * blockDim.x + threadIdx.x; i < M;
       i += (long long)gridDim.x * blockDim.x) {
    float v = xb[i];
    s1 += v;
    s2 += v * v;
  }
  __shared__ float a1[256], a2[256];
  a1[threadIdx.x] = s1; a2[threadIdx.x] = s2;
  __syncthreads();
  for (int s = 128; s > 0; s >>= 1) {
    if (threadIdx.x < s) { a1[threadIdx.x] += a1[threadIdx.x + s]; a2[threadIdx.x] += a2[threadIdx.x + s]; }
    __syncthreads();
  }
  if (threadIdx.x == 0) {
    atomicAdd(&stats[b * 2],     a1[0]);
    atomicAdd(&stats[b * 2 + 1], a2[0]);
  }
}

__global__ void ln_apply_kernel(float* __restrict__ xx, const float* __restrict__ stats,
                                const float* __restrict__ nw, const float* __restrict__ nb,
                                int Ti) {
  long long M = (long long)CC * NN * Ti;
  long long total = (long long)BB * M;
  long long idx = (long long)blockIdx.x * blockDim.x + threadIdx.x;
  if (idx >= total) return;
  int b = (int)(idx / M);
  long long r = idx % M;
  float mu  = stats[b * 2] / (float)M;
  float var = stats[b * 2 + 1] / (float)M - mu * mu;
  float inv = rsqrtf(var + EPSLN);
  xx[idx] = (xx[idx] - mu) * inv * nw[r] + nb[r];
}

// ---------------------------------------------------------------------------
// Head: relu(skip) -> end1(64->128)+relu -> end2(128->24)
// ---------------------------------------------------------------------------
__global__ void head_kernel(const float* __restrict__ skip,
                            const float* __restrict__ e1w, const float* __restrict__ e1b,
                            const float* __restrict__ e2w, const float* __restrict__ e2b,
                            float* __restrict__ out) {
  int idx = blockIdx.x * blockDim.x + threadIdx.x;
  if (idx >= BB * PRED * NN) return;
  int n = idx % NN;
  int r = idx / NN;
  int p = r % PRED;
  int b = r / PRED;
  float acc = e2b[p];
  for (int o2 = 0; o2 < CEND; ++o2) {
    float a1 = e1b[o2];
    for (int c = 0; c < CSKIP; ++c) {
      float sv = skip[((long long)b * CSKIP + c) * NN + n];
      sv = sv > 0.f ? sv : 0.f;
      a1 += e1w[o2 * CSKIP + c] * sv;
    }
    a1 = a1 > 0.f ? a1 : 0.f;
    acc += e2w[p * CEND + o2] * a1;
  }
  out[idx] = acc;  // idx == (b*PRED + p)*NN + n
}

// ---------------------------------------------------------------------------
// Host orchestration
// ---------------------------------------------------------------------------
static inline unsigned nblk(long long n) { return (unsigned)((n + 255) / 256); }

extern "C" void kernel_launch(void* const* d_in, const int* in_sizes, int n_in,
                              void* d_out, int out_size, void* d_ws, size_t ws_size,
                              hipStream_t stream) {
  (void)in_sizes; (void)n_in; (void)out_size; (void)ws_size;
  auto F = [&](int i) { return (const float*)d_in[i]; };

  const float* x      = F(0);
  const float* emb1   = F(1);
  const float* emb2   = F(2);
  const float* lin1w  = F(3);
  const float* lin1b  = F(4);
  const float* lin2w  = F(5);
  const float* lin2b  = F(6);
  const float* startw = F(7);
  const float* startb = F(8);
  const float* pos    = F(9);
  const float* skip0w = F(10);
  const float* skip0b = F(11);
  const float* skipEw = F(12);
  const float* skipEb = F(13);
  const float* end1w  = F(14);
  const float* end1b  = F(15);
  const float* end2w  = F(16);
  const float* end2b  = F(17);
  // layer i base: 18 + 24*i
  auto L = [&](int i, int k) { return (const float*)d_in[18 + 24 * i + k]; };

  float* out_main = (float*)d_out;                        // [B,24,N,1]
  float* dense    = out_main + (long long)BB * PRED * NN; // [N,N] dense adjacency

  // ---- workspace carve ----
  char* wp = (char*)d_ws;
  auto allocf = [&](long long n) { float* p = (float*)wp; wp += ((n * 4 + 255) / 256) * 256; return p; };
  auto alloch = [&](long long n) { f16* p = (f16*)wp;   wp += ((n * 2 + 255) / 256) * 256; return p; };

  const long long MAXE  = (long long)BB * CC * NN * TT;   // 44,040,192
  const long long MAXEP = (long long)BB * CC * NN * 192;  // padded-T (mult of 32)

  float* nv1    = allocf(NN * ND);
  float* nv2    = allocf(NN * ND);
  float* adp    = allocf(NN * NN);
  float* rowsum = allocf(NN);
  float* colsum = allocf(NN);
  f16*   Afwd   = alloch(NN * NN);
  f16*   Abwd   = alloch(NN * NN);
  float* skip   = allocf((long long)BB * CSKIP * NN);
  float* stats  = allocf(2 * BB);
  float* act0   = allocf(MAXE);
  float* act1   = allocf(MAXE);
  float* xconv  = allocf(MAXE);
  float* h1f    = allocf(MAXE);
  float* h2f    = allocf(MAXE);
  float* h1b    = allocf(MAXE);
  float* h2b    = allocf(MAXE);
  float* Yg     = allocf(MAXEP);
  f16*   Hh     = alloch(MAXEP);

  // ---- graph learning ----
  nodevec_kernel<<<NN, 64, 0, stream>>>(emb1, lin1w, lin1b, nv1);
  nodevec_kernel<<<NN, 64, 0, stream>>>(emb2, lin2w, lin2b, nv2);
  amat_kernel<<<nblk((long long)NN * NN), 256, 0, stream>>>(nv1, nv2, dense);
  topk_kernel<<<NN, 256, 0, stream>>>(dense, adp);
  adj_sums_kernel<<<NN, 256, 0, stream>>>(adp, rowsum, 0);
  adj_sums_kernel<<<NN, 256, 0, stream>>>(adp, colsum, 1);
  normadj_f16_kernel<<<nblk((long long)NN * NN), 256, 0, stream>>>(adp, rowsum, colsum, Afwd, Abwd);

  // ---- front end ----
  start_kernel<<<nblk(MAXE), 256, 0, stream>>>(x, startw, startb, pos, act0);
  skip0_kernel<<<nblk((long long)BB * CSKIP * NN), 256, 0, stream>>>(x, skip0w, skip0b, skip);

  float* cur = act0;
  float* nxt = act1;
  int Tprev = TT;
  const int TL[3] = {162, 150, 126};

  for (int i = 0; i < 3; ++i) {
    const int Ti = TL[i];
    const int dil = 1 << i;
    const int Tp = (Ti + 31) / 32 * 32;  // 192, 160, 128
    const long long nTi = (long long)BB * CC * NN * Ti;
    const long long nTp = (long long)BB * CC * NN * Tp;

    IncArgs ia;
    for (int k = 0; k < 4; ++k) {
      ia.fw[k] = L(i, k);      ia.fb[k] = L(i, 4 + k);
      ia.gw[k] = L(i, 8 + k);  ia.gb[k] = L(i, 12 + k);
    }
    inception_kernel<<<nblk(nTi), 256, 0, stream>>>(cur, xconv, ia, Tprev, Ti, dil);
    skipconv_kernel<<<nblk((long long)BB * CSKIP * NN), 256, 0, stream>>>(
        xconv, L(i, 16), L(i, 17), skip, Ti);

    dim3 ggrid((unsigned)(Tp / 32), 4u, (unsigned)(BB * CC));
    // forward-direction hops
    tof16_pad_kernel<<<nblk(nTp), 256, 0, stream>>>(xconv, Hh, Ti, Tp);
    nconv_wmma_kernel<<<ggrid, 256, 0, stream>>>(Afwd, Hh, Yg, Tp);
    hop_combine_kernel<<<nblk(nTp), 256, 0, stream>>>(xconv, Yg, h1f, Hh, Ti, Tp);
    nconv_wmma_kernel<<<ggrid, 256, 0, stream>>>(Afwd, Hh, Yg, Tp);
    hop_combine_kernel<<<nblk(nTp), 256, 0, stream>>>(xconv, Yg, h2f, Hh, Ti, Tp);
    // backward-direction hops
    tof16_pad_kernel<<<nblk(nTp), 256, 0, stream>>>(xconv, Hh, Ti, Tp);
    nconv_wmma_kernel<<<ggrid, 256, 0, stream>>>(Abwd, Hh, Yg, Tp);
    hop_combine_kernel<<<nblk(nTp), 256, 0, stream>>>(xconv, Yg, h1b, Hh, Ti, Tp);
    nconv_wmma_kernel<<<ggrid, 256, 0, stream>>>(Abwd, Hh, Yg, Tp);
    hop_combine_kernel<<<nblk(nTp), 256, 0, stream>>>(xconv, Yg, h2b, Hh, Ti, Tp);

    mixmlp_kernel<<<nblk(nTi), 256, 0, stream>>>(
        xconv, h1f, h2f, h1b, h2b,
        L(i, 18), L(i, 19), L(i, 20), L(i, 21),
        cur, Tprev, Ti, nxt);

    zero_kernel<<<1, 64, 0, stream>>>(stats, 2 * BB);
    ln_reduce_kernel<<<dim3(1024, BB), 256, 0, stream>>>(nxt, stats, Ti);
    ln_apply_kernel<<<nblk(nTi), 256, 0, stream>>>(nxt, stats, L(i, 22), L(i, 23), Ti);

    float* tmp = cur; cur = nxt; nxt = tmp;
    Tprev = Ti;
  }

  // ---- tail ----
  skipconv_kernel<<<nblk((long long)BB * CSKIP * NN), 256, 0, stream>>>(
      cur, skipEw, skipEb, skip, TL[2]);
  head_kernel<<<nblk((long long)BB * PRED * NN), 256, 0, stream>>>(
      skip, end1w, end1b, end2w, end2b, out_main);
}
